// AttentionHead_4243427688754
// MI455X (gfx1250) — compile-verified, matrix operand order checked by reading, not run on hardware
//
#include <hip/hip_runtime.h>
#include <hip/hip_bf16.h>
#include <math.h>

// ---------------------------------------------------------------------------
// AttentionHead (B=4, T=4096, D=1024, H=64), causal softmax attention.
// bf16 WMMA pipeline:
//   wtranspose  : weights -> bf16 W^T (D**-0.5 * log2(e) folded into Wq)
//   qkv_proj    : TDM (tensor_load_to_lds) stages x row-tile; WMMA GEMM
//   flash_attn  : async global->LDS double-buffered K/V; WMMA flash attention
//                 with exp2-domain online softmax (v_exp_f32 is exp2 natively)
// ---------------------------------------------------------------------------

typedef __bf16 bf16_t;
typedef __attribute__((ext_vector_type(16))) __bf16 v16bf;
typedef __attribute__((ext_vector_type(8)))  __bf16 v8bf;
typedef __attribute__((ext_vector_type(8)))  float  v8f;
typedef __attribute__((ext_vector_type(4)))  unsigned u32x4;
typedef __attribute__((ext_vector_type(4)))  int      i32x4;
typedef __attribute__((ext_vector_type(8)))  int      i32x8;

#define WMMA_BF16(A, B, C) \
    __builtin_amdgcn_wmma_f32_16x16x32_bf16(false, (A), false, (B), (short)0, (C), false, false)

static constexpr int Bsz = 4;
static constexpr int T   = 4096;
static constexpr int D   = 1024;
static constexpr int H   = 64;

// Build a 16-element bf16 WMMA operand from two contiguous 8-element halves
// (K groups {k0..k0+7} and {k0+16..k0+23} of the ISA 7.12.2 16-bit layouts).
__device__ inline v16bf load_bf16_operand(const bf16_t* p) {
    v8bf lo = *(const v8bf*)p;
    v8bf hi = *(const v8bf*)(p + 16);
    return __builtin_shufflevector(lo, hi,
           0, 1, 2, 3, 4, 5, 6, 7, 8, 9, 10, 11, 12, 13, 14, 15);
}

// ---------------------------------------------------------------------------
// Kernel 0: transpose weights to bf16 W^T[H x D].
// Wq additionally absorbs D**-0.5 * log2(e) so attention scores are produced
// directly in the exp2 domain (hardware v_exp_f32 computes 2^x).
// ---------------------------------------------------------------------------
__global__ __launch_bounds__(256) void wtranspose(
    const float* __restrict__ Wq, const float* __restrict__ Wk,
    const float* __restrict__ Wv,
    bf16_t* __restrict__ Wtq, bf16_t* __restrict__ Wtk, bf16_t* __restrict__ Wtv)
{
    const int idx = blockIdx.x * 256 + threadIdx.x;   // over H*D
    const int n = idx / D;
    const int k = idx % D;
    const size_t src = (size_t)k * H + n;
    const float qscale = 0.03125f * 1.44269504088896340736f;  // D^-0.5 * log2(e)
    Wtq[idx] = (bf16_t)(Wq[src] * qscale);
    Wtk[idx] = (bf16_t)Wk[src];
    Wtv[idx] = (bf16_t)Wv[src];
}

// ---------------------------------------------------------------------------
// Kernel 1: Q/K/V projection.  grid = B*T/16 blocks, 128 threads (4 waves).
// The Tensor Data Mover DMAs the whole 16x1024 fp32 x row-tile into LDS with
// ONE descriptor (D# per ISA 8.3/8.4); after a single barrier the k-loop is
// barrier-free: A tiles via ds_load_b128 (+cvt to bf16), B tiles via
// contiguous b128 loads from W^T, 3 WMMAs per 32-deep chunk (unroll 4).
// V is written transposed (Vt[B,H,T]) with one packed b128 store per lane.
// ---------------------------------------------------------------------------
__global__ __launch_bounds__(128) void qkv_proj(
    const float* __restrict__ x,
    const bf16_t* __restrict__ Wtq, const bf16_t* __restrict__ Wtk,
    const bf16_t* __restrict__ Wtv,
    bf16_t* __restrict__ Qb, bf16_t* __restrict__ Kb, bf16_t* __restrict__ Vt)
{
    __shared__ __attribute__((aligned(16))) float xs[16 * D];   // 64 KB row tile

    const int wid  = threadIdx.x >> 5;
    const int lane = threadIdx.x & 31;
    const int half = lane >> 4;
    const int mrow = lane & 15;
    const int k0   = half * 8;            // K-group base for this half-wave
    const int r0   = blockIdx.x * 16;     // row tile in [0, B*T)
    const int n0   = wid * 16;            // column tile within H

    // ---- TDM: one descriptor moves rows [r0, r0+16) x [0, D) of x into LDS.
    if (wid == 0) {
        const unsigned lds0 = (unsigned)(size_t)(void*)xs;  // ptrtoint: escapes xs
        const unsigned long long ga =
            (unsigned long long)(size_t)(x + (size_t)r0 * D);
        u32x4 g0;
        g0[0] = 1u;                                          // count=1 (valid D#)
        g0[1] = lds0;                                        // lds_addr
        g0[2] = (unsigned)ga;                                // global_addr[31:0]
        g0[3] = (unsigned)((ga >> 32) & 0x01FFFFFFu)         // global_addr[56:32]
              | (2u << 30);                                  // type=2 ("image")
        i32x8 g1 = {};
        g1[0] = (int)(2u << 16);                             // data_size=2 -> 4B
        g1[1] = (int)((unsigned)(D & 0xFFFF) << 16);         // tensor_dim0 lo16
        g1[2] = (int)(((unsigned)D >> 16)                    // tensor_dim0 hi16
              | ((unsigned)((Bsz * T) & 0xFFFF) << 16));     // tensor_dim1 lo16
        g1[3] = (int)(((unsigned)(Bsz * T) >> 16)            // tensor_dim1 hi16
              | ((unsigned)D << 16));                        // tile_dim0 = 1024
        g1[4] = 16;                                          // tile_dim1 = 16
        g1[5] = D;                                           // tensor_dim0_stride
        g1[6] = 0;
        g1[7] = 0;
        i32x4 gz = {};
#if __has_include(<hip/amd_detail/amd_gfx1250_TDM.h>)
        i32x8 gz8 = {};
        __builtin_amdgcn_tensor_load_to_lds(g0, g1, gz, gz, gz8, 0);
#else
        __builtin_amdgcn_tensor_load_to_lds(g0, g1, gz, gz, 0);
#endif
        __builtin_amdgcn_s_wait_tensorcnt((short)0);
    }
    __syncthreads();

    v8f accq = {}, acck = {}, accv = {};

    const bf16_t* wq = Wtq + (size_t)(n0 + mrow) * D;
    const bf16_t* wk = Wtk + (size_t)(n0 + mrow) * D;
    const bf16_t* wv = Wtv + (size_t)(n0 + mrow) * D;

#pragma unroll 4
    for (int kc = 0; kc < D / 32; ++kc) {
        // ---- A tile: LDS fp32 -> bf16 in WMMA A layout (4x ds_load_b128)
        const float* xrow = xs + mrow * D + kc * 32 + k0;
        const float4* xr  = (const float4*)xrow;
        const float4* xr2 = (const float4*)(xrow + 16);
        float4 fa = xr[0],  fb = xr[1];
        float4 fc = xr2[0], fd = xr2[1];
        float fv[16] = { fa.x, fa.y, fa.z, fa.w, fb.x, fb.y, fb.z, fb.w,
                         fc.x, fc.y, fc.z, fc.w, fd.x, fd.y, fd.z, fd.w };
        v16bf a;
#pragma unroll
        for (int j = 0; j < 16; ++j) a[j] = (bf16_t)fv[j];

        // ---- B tiles from transposed weights: contiguous b128 pairs
        const size_t wo = (size_t)kc * 32 + k0;
        v16bf bq = load_bf16_operand(wq + wo);
        v16bf bk = load_bf16_operand(wk + wo);
        v16bf bv = load_bf16_operand(wv + wo);

        accq = WMMA_BF16(a, bq, accq);
        acck = WMMA_BF16(a, bk, acck);
        accv = WMMA_BF16(a, bv, accv);
    }

    // C layout: VGPR p, lanes 0-15 -> M=p ; lanes 16-31 -> M=p+8 ; N = lane&15.
    const int b      = r0 / T;
    const int rowInB = r0 - b * T;
#pragma unroll
    for (int p = 0; p < 8; ++p) {
        const int row = r0 + p + 8 * half;
        const size_t o = (size_t)row * H + n0 + mrow;
        Qb[o] = (bf16_t)accq[p];          // scales already folded into Wtq
        Kb[o] = (bf16_t)acck[p];
    }
    v8bf vv;                               // transposed V: rows contiguous in p
#pragma unroll
    for (int p = 0; p < 8; ++p) vv[p] = (bf16_t)accv[p];
    *(v8bf*)(Vt + ((size_t)b * H + n0 + mrow) * T + rowInB + 8 * half) = vv;
}

// ---------------------------------------------------------------------------
// Kernel 2: causal flash attention.  grid = B*T/(16*4) blocks, 128 threads.
// One wave owns a 16-row query tile and the 16x64 fp32 accumulator.
// K/V key blocks (32 keys) are double-buffered per wave in LDS via
// global_load_async_to_lds_b128 (issue before compute, s_wait_asynccnt after)
// so the next block's fetch hides under this block's 8 WMMAs + softmax.
// Softmax runs entirely in the exp2 domain (log2(e) pre-folded into Q).
// ---------------------------------------------------------------------------
__global__ __launch_bounds__(128) void flash_attn(
    const bf16_t* __restrict__ Qb,
    const bf16_t* __restrict__ Kb,
    const bf16_t* __restrict__ Vt,
    float* __restrict__ out)
{
    __shared__ __attribute__((aligned(16))) bf16_t pl[4][16 * 32];     // P relayout
    __shared__ __attribute__((aligned(16))) bf16_t kvs[4][2][4096];    // [wave][buf][K|V]

    const int wid  = threadIdx.x >> 5;
    const int lane = threadIdx.x & 31;
    const int half = lane >> 4;
    const int mrow = lane & 15;
    const int k0   = half * 8;

    const int qt = blockIdx.x * 4 + wid;     // global 16-row query tile id
    const int b  = qt >> 8;                  // 256 tiles per batch
    const int r0 = (qt & 255) * 16;

    const bf16_t* Qbase = Qb + ((size_t)b * T + r0) * H;
    const bf16_t* Kbase = Kb + (size_t)b * T * H;
    const bf16_t* Vbase = Vt + (size_t)b * H * T;

    // Async-stage one 32-key block: K rows [s0,s0+32) x H (contiguous 4KB) and
    // Vt rows [0,H) x cols [s0,s0+32) (64B segments) into this wave's buffer.
    auto stage_kv = [&](int s0, int bufi) {
        const bf16_t* kg = Kbase + (size_t)s0 * H + lane * 8;
        bf16_t* kl = &kvs[wid][bufi][lane * 8];
#pragma unroll
        for (int j = 0; j < 8; ++j) {
            const unsigned dst = (unsigned)(size_t)(void*)(kl + j * 256);
            const bf16_t* src = kg + j * 256;
            asm volatile("global_load_async_to_lds_b128 %0, %1, off"
                         :: "v"(dst), "v"(src) : "memory");
        }
#pragma unroll
        for (int j = 0; j < 8; ++j) {
            const int c   = j * 32 + lane;          // 16-byte chunk id (256 total)
            const int row = c >> 2, seg = c & 3;    // Vt row, 8-elem segment
            const bf16_t* src = Vbase + (size_t)row * T + s0 + seg * 8;
            const unsigned dst =
                (unsigned)(size_t)(void*)&kvs[wid][bufi][2048 + c * 8];
            asm volatile("global_load_async_to_lds_b128 %0, %1, off"
                         :: "v"(dst), "v"(src) : "memory");
        }
    };

    // Q in A layout for both 32-wide h-chunks (direct b128 global loads).
    v16bf aq[2];
#pragma unroll
    for (int hc = 0; hc < 2; ++hc)
        aq[hc] = load_bf16_operand(Qbase + (size_t)mrow * H + hc * 32 + k0);

    v8f acc[4] = {};
    float mrun[8], lrun[8];
#pragma unroll
    for (int p = 0; p < 8; ++p) { mrun[p] = -1e30f; lrun[p] = 0.0f; }

    const int nblk = (r0 + 47) >> 5;         // causal: cover columns <= r0+15
    stage_kv(0, 0);
    asm volatile("s_wait_asynccnt 0x0" ::: "memory");

    int buf = 0;
    for (int blk = 0; blk < nblk; ++blk) {
        const int s0 = blk * 32;
        if (blk + 1 < nblk) {
            stage_kv(s0 + 32, buf ^ 1);      // overlap next block with compute
            __builtin_prefetch(Kbase + (size_t)(s0 + 64) * H, 0, 1);
        }
        const bf16_t* kbuf = &kvs[wid][buf][0];
        const bf16_t* vbuf = &kvs[wid][buf][2048];

        // ---- S = Q * K^T (exp2 domain), two 16x16 tiles over this key block
        v8f st[2];
#pragma unroll
        for (int tile = 0; tile < 2; ++tile) {
            const int s = s0 + tile * 16 + mrow;     // key row = S column
            v8f sacc = {};
#pragma unroll
            for (int hc = 0; hc < 2; ++hc) {
                v16bf bK = load_bf16_operand(
                    kbuf + (size_t)(tile * 16 + mrow) * H + hc * 32 + k0);
                sacc = WMMA_BF16(aq[hc], bK, sacc);
            }
#pragma unroll
            for (int p = 0; p < 8; ++p) {
                const int row = r0 + p + 8 * half;
                sacc[p] = (s <= row) ? sacc[p] : -1e30f;   // causal mask
            }
            st[tile] = sacc;
        }

        // ---- online softmax update, exp2 domain (v_exp_f32 == 2^x)
        float alpha[8];
#pragma unroll
        for (int p = 0; p < 8; ++p) {
            float m = fmaxf(st[0][p], st[1][p]);
#pragma unroll
            for (int off = 1; off < 16; off <<= 1)
                m = fmaxf(m, __shfl_xor(m, off, 16));
            const float mn = fmaxf(mrun[p], m);
            alpha[p] = exp2f(mrun[p] - mn);
            const float e0 = exp2f(st[0][p] - mn);
            const float e1 = exp2f(st[1][p] - mn);
            st[0][p] = e0;
            st[1][p] = e1;
            float rs = e0 + e1;
#pragma unroll
            for (int off = 1; off < 16; off <<= 1)
                rs += __shfl_xor(rs, off, 16);
            lrun[p] = lrun[p] * alpha[p] + rs;
            mrun[p] = mn;
        }
#pragma unroll
        for (int t2 = 0; t2 < 4; ++t2)
#pragma unroll
            for (int p = 0; p < 8; ++p) acc[t2][p] *= alpha[p];

        // ---- P: C layout -> row-major bf16 in LDS -> reload in A layout
#pragma unroll
        for (int tile = 0; tile < 2; ++tile)
#pragma unroll
            for (int p = 0; p < 8; ++p)
                pl[wid][(p + 8 * half) * 32 + tile * 16 + mrow] = (bf16_t)st[tile][p];
        asm volatile("s_wait_dscnt 0x0" ::: "memory");
        v16bf aP = load_bf16_operand(&pl[wid][mrow * 32 + k0]);

        // ---- O += P * V  (B tiles from staged Vt block, ds_load_b128 pairs)
#pragma unroll
        for (int t2 = 0; t2 < 4; ++t2) {
            v16bf bV = load_bf16_operand(
                vbuf + (size_t)(t2 * 16 + mrow) * 32 + k0);
            acc[t2] = WMMA_BF16(aP, bV, acc[t2]);
        }

        asm volatile("s_wait_asynccnt 0x0" ::: "memory");  // next buffer ready
        buf ^= 1;
    }

    // ---- finalize: divide by softmax denominator, store fp32 output
#pragma unroll
    for (int t2 = 0; t2 < 4; ++t2)
#pragma unroll
        for (int p = 0; p < 8; ++p) {
            const int row = r0 + p + 8 * half;
            out[((size_t)b * T + row) * H + t2 * 16 + mrow] = acc[t2][p] / lrun[p];
        }
}

// ---------------------------------------------------------------------------
// Host launcher
// ---------------------------------------------------------------------------
extern "C" void kernel_launch(void* const* d_in, const int* in_sizes, int n_in,
                              void* d_out, int out_size, void* d_ws, size_t ws_size,
                              hipStream_t stream)
{
    const float* x  = (const float*)d_in[0];
    const float* Wq = (const float*)d_in[1];
    const float* Wk = (const float*)d_in[2];
    const float* Wv = (const float*)d_in[3];
    float* out = (float*)d_out;

    const size_t nQ = (size_t)Bsz * T * H;   // elements per Q/K/V
    const size_t nW = (size_t)H * D;         // elements per transposed weight
    if (ws_size < (3 * nQ + 3 * nW) * sizeof(bf16_t)) return;  // ~6.4 MiB

    bf16_t* Qb  = (bf16_t*)d_ws;
    bf16_t* Kb  = Qb + nQ;
    bf16_t* Vt  = Kb + nQ;
    bf16_t* Wtq = Vt + nQ;
    bf16_t* Wtk = Wtq + nW;
    bf16_t* Wtv = Wtk + nW;

    wtranspose<<<(H * D) / 256, 256, 0, stream>>>(Wq, Wk, Wv, Wtq, Wtk, Wtv);
    qkv_proj<<<(Bsz * T) / 16, 128, 0, stream>>>(x, Wtq, Wtk, Wtv, Qb, Kb, Vt);
    flash_attn<<<(Bsz * T) / (16 * 4), 128, 0, stream>>>(Qb, Kb, Vt, out);
}
